// YAADLayer_91130616086884
// MI455X (gfx1250) — compile-verified, matrix operand order checked by reading, not run on hardware
//
#include <hip/hip_runtime.h>
#include <hip/hip_bf16.h>
#include <math.h>

typedef __attribute__((ext_vector_type(16))) _Float16 v16h;
typedef __attribute__((ext_vector_type(8)))  float    v8f;
typedef __attribute__((ext_vector_type(4)))  unsigned int v4u;
typedef __attribute__((ext_vector_type(8)))  int      v8i;
typedef __attribute__((ext_vector_type(4)))  int      v4i;

#if defined(__has_builtin)
#if __has_builtin(__builtin_amdgcn_tensor_load_to_lds) && __has_builtin(__builtin_amdgcn_s_wait_tensorcnt)
#define USE_TDM 1
#endif
#endif

#define Bsz   4
#define Sseq  4096
#define Ddim  1024
#define Hdim  128
#define Mdim  256
#define Rdim  64
#define CH    64
#define NCHK  64
#define ROWS  (Bsz*Sseq)   // 16384

#define TILE_M 128
#define TILE_N 128
#define TILE_K 32

#ifdef USE_TDM
// Issue one TDM load of a (128 rows x 32 elems) f16 tile into LDS.
// D# per CDNA5 ISA ch.8: group0 = {count/lds_addr/global_addr/type},
// group1 = {mask, data_size, dims, tile dims, strides}; groups 2/3 zero (2D).
// 6-arg builtin form (clang-23 / therock-10.0 headers).
__device__ __forceinline__ void tdm_load_tile(unsigned lds_byte_off,
                                              const _Float16* gsrc, int K)
{
    unsigned long long ga = (unsigned long long)(size_t)gsrc;
    v4u g0;
    g0[0] = 1u;                                     // count=1 (valid user desc)
    g0[1] = lds_byte_off;                           // lds_addr[31:0]
    g0[2] = (unsigned)(ga & 0xFFFFFFFFu);           // global_addr[31:0]
    g0[3] = (unsigned)((ga >> 32) & 0x01FFFFFFu)    // global_addr[56:32]
          | (2u << 30);                             // type=2 ("image")
    const unsigned td0 = (unsigned)K;               // tensor_dim0 (elements)
    const unsigned td1 = (unsigned)TILE_M;          // tensor_dim1 (rows)
    const unsigned long long st0 = (unsigned long long)K; // dim0 stride (elems)
    v8i g1;
    g1[0] = (int)(1u << 16);                        // wg_mask=0, data_size=1 (2B)
    g1[1] = (int)((td0 & 0xFFFFu) << 16);           // abar_addr=0 | tensor_dim0 lo
    g1[2] = (int)((td0 >> 16) | ((td1 & 0xFFFFu) << 16)); // td0 hi | td1 lo
    g1[3] = (int)((td1 >> 16) | ((unsigned)TILE_K << 16)); // td1 hi | tile_dim0=32
    g1[4] = (int)((unsigned)TILE_M);                // tile_dim1=128, tile_dim2=0
    g1[5] = (int)(st0 & 0xFFFFFFFFu);               // tensor_dim0_stride lo
    g1[6] = (int)((st0 >> 32) & 0xFFFFu);           // stride hi | dim1_stride lo=0
    g1[7] = 0;                                      // dim1_stride hi = 0
    v4i z4 = {0, 0, 0, 0};
    v8i z8 = {0, 0, 0, 0, 0, 0, 0, 0};
    __builtin_amdgcn_tensor_load_to_lds(g0, g1, z4, z4, z8, 0);
}
#endif

// ---------------------------------------------------------------------------
// f16 GEMM, fp32 accumulate, via v_wmma_f32_16x16x32_f16 (wave32).
// A: M x K row-major (K-contiguous).  BT: N x K row-major (K-contiguous).
// Both LDS tiles share the K-contiguous layout -> every fragment is two
// contiguous 16B LDS reads per lane (ds_load_b128), no repacking.
// Staging: TDM (tensor_load_to_lds, wave0=A / wave1=B, TENSORcnt ping-pong)
// when available; else register double-buffer + global_prefetch.
// Block: 256 threads = 8 waves, wave grid 4(M) x 2(N); each wave 32x64 C tile.
// Optional fused epilogue: C[i] = acc * sigmoid(C[i])  (gate pre-stored in C).
// ---------------------------------------------------------------------------
__global__ __launch_bounds__(256)
void gemm_f16_wmma(const _Float16* __restrict__ A, const _Float16* __restrict__ BT,
                   float* __restrict__ C, int M, int N, int K, int fuse_sigmoid)
{
    __shared__ __align__(16) _Float16 As[2][TILE_M][TILE_K];  // 2 x 8KB
    __shared__ __align__(16) _Float16 Bs[2][TILE_N][TILE_K];  // 2 x 8KB

    const int tid  = threadIdx.x;
    const int lane = tid & 31;
    const int wave = tid >> 5;
    const int wm   = (wave & 3) * 32;   // wave offset in M within tile
    const int wn   = (wave >> 2) * 64;  // wave offset in N within tile
    const int bm   = blockIdx.y * TILE_M;
    const int bn   = blockIdx.x * TILE_N;
    const int ml   = lane & 15;
    const int kh   = (lane >> 4) * 8;   // K half select per 16-bit A/B layout

    v8f acc[2][4];
    #pragma unroll
    for (int i = 0; i < 2; i++)
        #pragma unroll
        for (int j = 0; j < 4; j++)
            #pragma unroll
            for (int r = 0; r < 8; r++) acc[i][j][r] = 0.0f;

    const int nk = K / TILE_K;

    // Per-K-step compute: fragments (two b128 LDS reads each) + 8 WMMAs.
    auto compute_tile = [&](int buf) {
        v16h af[2], bf[4];
        #pragma unroll
        for (int i = 0; i < 2; i++) {
            const int row = wm + i * 16 + ml;
            #pragma unroll
            for (int j = 0; j < 8; j++) {
                af[i][j]     = As[buf][row][kh + j];
                af[i][8 + j] = As[buf][row][16 + kh + j];
            }
        }
        #pragma unroll
        for (int j2 = 0; j2 < 4; j2++) {
            const int col = wn + j2 * 16 + ml;
            #pragma unroll
            for (int j = 0; j < 8; j++) {
                bf[j2][j]     = Bs[buf][col][kh + j];
                bf[j2][8 + j] = Bs[buf][col][16 + kh + j];
            }
        }
        #pragma unroll
        for (int i = 0; i < 2; i++)
            #pragma unroll
            for (int j2 = 0; j2 < 4; j2++)
                acc[i][j2] = __builtin_amdgcn_wmma_f32_16x16x32_f16(
                    false, af[i], false, bf[j2], (short)0, acc[i][j2], false, false);
    };

#ifdef USE_TDM
    // wave0 DMA-stages A tiles, wave1 stages B tiles; TENSORcnt is per-wave
    // and in-order, so waiting <=1 guarantees the older tile has landed.
    const _Float16* tbase =
        (wave == 0) ? (A + (size_t)bm * K) : (BT + (size_t)bn * K);
    const unsigned ldsbase =
        (wave == 0) ? (unsigned)(size_t)&As[0][0][0]
                    : (unsigned)(size_t)&Bs[0][0][0];
    const unsigned bufstride = (unsigned)(TILE_M * TILE_K * sizeof(_Float16));

    if (wave < 2) tdm_load_tile(ldsbase, tbase, K);

    for (int t = 0; t < nk; t++) {
        const int buf = t & 1;
        if (wave < 2) {
            if (t + 1 < nk) {
                tdm_load_tile(ldsbase + (unsigned)((t + 1) & 1) * bufstride,
                              tbase + (size_t)(t + 1) * TILE_K, K);
                __builtin_amdgcn_s_wait_tensorcnt(1);   // tile t landed
            } else {
                __builtin_amdgcn_s_wait_tensorcnt(0);
            }
        }
        __syncthreads();          // publish LDS tile to all waves
        compute_tile(buf);
        __syncthreads();          // tile consumed; safe to overwrite at t+2
    }
#else
    // Fallback: register-staged global loads, double-buffered LDS,
    // one barrier per K-step, +2-tile global_prefetch.
    const int sr = tid >> 1;            // 0..127
    const int sc = (tid & 1) * 16;      // 0 or 16
    const _Float16* aptr = A  + (size_t)(bm + sr) * K + sc;
    const _Float16* bptr = BT + (size_t)(bn + sr) * K + sc;

    uint4 a0 = *(const uint4*)(aptr);
    uint4 a1 = *(const uint4*)(aptr + 8);
    uint4 b0 = *(const uint4*)(bptr);
    uint4 b1 = *(const uint4*)(bptr + 8);

    for (int t = 0; t < nk; t++) {
        const int buf = t & 1;
        *(uint4*)(&As[buf][sr][sc])     = a0;
        *(uint4*)(&As[buf][sr][sc + 8]) = a1;
        *(uint4*)(&Bs[buf][sr][sc])     = b0;
        *(uint4*)(&Bs[buf][sr][sc + 8]) = b1;
        __syncthreads();
        if (t + 1 < nk) {
            const _Float16* an  = aptr + (size_t)(t + 1) * TILE_K;
            const _Float16* bn2 = bptr + (size_t)(t + 1) * TILE_K;
            a0 = *(const uint4*)(an);  a1 = *(const uint4*)(an + 8);
            b0 = *(const uint4*)(bn2); b1 = *(const uint4*)(bn2 + 8);
        }
        if (t + 2 < nk) {
            __builtin_prefetch(aptr + (size_t)(t + 2) * TILE_K, 0, 1);
            __builtin_prefetch(bptr + (size_t)(t + 2) * TILE_K, 0, 1);
        }
        compute_tile(buf);
    }
#endif

    // C/D layout: VGPR r -> M = r (lanes 0-15) / 8+r (lanes 16-31); N = lane&15
    #pragma unroll
    for (int i = 0; i < 2; i++) {
        #pragma unroll
        for (int j2 = 0; j2 < 4; j2++) {
            #pragma unroll
            for (int r = 0; r < 8; r++) {
                const int row = bm + wm + i * 16 + (lane >> 4) * 8 + r;
                const int col = bn + wn + j2 * 16 + ml;
                const size_t off = (size_t)row * N + col;
                float v = acc[i][j2][r];
                if (fuse_sigmoid) {
                    float g = C[off];
                    v *= 1.0f / (1.0f + expf(-g));
                }
                C[off] = v;
            }
        }
    }
}

// ---------------------------------------------------------------------------
// Elementwise / packing kernels (all B matrices emitted TRANSPOSED: N x K)
// ---------------------------------------------------------------------------
__global__ void cvt_f32_f16(const float* __restrict__ in, _Float16* __restrict__ out, int n)
{
    int i = blockIdx.x * blockDim.x + threadIdx.x;
    if (i < n) out[i] = (_Float16)in[i];
}

// out (N x K) = transpose of in (K x N), cast to f16
__global__ void cvt_f32_f16_t(const float* __restrict__ in, _Float16* __restrict__ out,
                              int K, int N)
{
    int i = blockIdx.x * blockDim.x + threadIdx.x;
    if (i >= K * N) return;
    int n = i / K, k = i - n * K;
    out[i] = (_Float16)in[(size_t)k * N + n];
}

__global__ void pack_kv_t(const float* __restrict__ Wk, const float* __restrict__ Wv,
                          _Float16* __restrict__ out)   // 256 x 1024 (N x K)
{
    int i = blockIdx.x * blockDim.x + threadIdx.x;
    if (i >= 256 * Ddim) return;
    int n = i >> 10, k = i & 1023;
    float v = (n < 128) ? Wk[(size_t)k * Hdim + n] : Wv[(size_t)k * Hdim + (n - 128)];
    out[i] = (_Float16)v;
}

__global__ void pack_p1_t(const float* __restrict__ We1, const float* __restrict__ Wd1,
                          const float* __restrict__ Wa1, _Float16* __restrict__ out) // 256 x 1024
{
    int i = blockIdx.x * blockDim.x + threadIdx.x;
    if (i >= 256 * Ddim) return;
    int n = i >> 10, k = i & 1023;
    float v = 0.0f;
    if (n < 64)       v = We1[(size_t)k * Rdim + n];
    else if (n < 128) v = Wd1[(size_t)k * Rdim + (n - 64)];
    else if (n < 192) v = Wa1[(size_t)k * Rdim + (n - 128)];
    out[i] = (_Float16)v;
}

__global__ void pack_diag_t(const float* __restrict__ We2, const float* __restrict__ Wd2,
                            const float* __restrict__ Wa2, _Float16* __restrict__ out) // 384 x 256
{
    int i = blockIdx.x * blockDim.x + threadIdx.x;
    if (i >= 384 * 256) return;
    int n = i >> 8, k = i & 255;
    float v = 0.0f;
    if (n < 128 && k < 64)                          v = We2[(size_t)k * Hdim + n];
    else if (n >= 128 && n < 256 && k >= 64 && k < 128)
                                                    v = Wd2[(size_t)(k - 64) * Hdim + (n - 128)];
    else if (n >= 256 && k >= 128 && k < 192)       v = Wa2[(size_t)(k - 128) * Hdim + (n - 256)];
    out[i] = (_Float16)v;
}

// eta (cols 0-127) / delta (128-255) softplus; alpha (256-383) sigmoid; in-place
__global__ void eda_act(float* __restrict__ eda, int n)
{
    int i = blockIdx.x * blockDim.x + threadIdx.x;
    if (i >= n) return;
    int col = i % 384;
    float x = eda[i];
    eda[i] = (col < 256) ? ((x > 20.0f) ? x : log1pf(expf(x)))
                         : (1.0f / (1.0f + expf(-x)));
}

// ---------------------------------------------------------------------------
// Causal depthwise conv(4) + LayerNorm for k, conv only for v.
// kv: (ROWS x 256) = [k_raw | v_raw]
// ---------------------------------------------------------------------------
__global__ __launch_bounds__(128)
void conv_ln(const float* __restrict__ kv, const float* __restrict__ cw,
             const float* __restrict__ cb, const float* __restrict__ kng,
             const float* __restrict__ knb,
             float* __restrict__ kpost, float* __restrict__ vpost)
{
    const int row = blockIdx.x;       // 0..ROWS-1
    const int h   = threadIdx.x;      // 0..127
    const int s   = row & (Sseq - 1); // position within batch

    float ka = cb[h], va = cb[h];
    #pragma unroll
    for (int j = 0; j < 4; j++) {
        int sj = s - 3 + j;
        if (sj >= 0) {
            const float* src = kv + (size_t)(row - (3 - j)) * 256;
            float w = cw[h * 4 + j];
            ka += w * src[h];
            va += w * src[128 + h];
        }
    }

    __shared__ float red[128];
    red[h] = ka; __syncthreads();
    for (int off = 64; off > 0; off >>= 1) { if (h < off) red[h] += red[h + off]; __syncthreads(); }
    float mean = red[0] * (1.0f / 128.0f);
    __syncthreads();
    float d = ka - mean;
    red[h] = d * d; __syncthreads();
    for (int off = 64; off > 0; off >>= 1) { if (h < off) red[h] += red[h + off]; __syncthreads(); }
    float var = red[0] * (1.0f / 128.0f);

    kpost[(size_t)row * Hdim + h] = d * rsqrtf(var + 1e-5f) * kng[h] + knb[h];
    vpost[(size_t)row * Hdim + h] = va;
}

// ---------------------------------------------------------------------------
// Sequential chunked scan: one block per batch, 64 chunks.
// State W1 (256x128), W2 (128x256) in global ws; biases in LDS.
// ---------------------------------------------------------------------------
__global__ __launch_bounds__(256)
void scan_kernel(const float* __restrict__ kpost, const float* __restrict__ vpost,
                 const float* __restrict__ eda,
                 const float* __restrict__ mW1, const float* __restrict__ mb1,
                 const float* __restrict__ mW2, const float* __restrict__ mb2,
                 float* __restrict__ W1state, float* __restrict__ W2state,
                 float* __restrict__ combined)
{
    const int b   = blockIdx.x;
    const int tid = threadIdx.x;
    float* W1s = W1state + (size_t)b * (Mdim * Hdim);
    float* W2s = W2state + (size_t)b * (Mdim * Hdim);

    __shared__ __align__(16) _Float16 act[CH][Mdim];   // 32KB
    __shared__ __align__(16) _Float16 errh[CH][Hdim];  // 16KB
    __shared__ float b1s[Mdim], b2s[Hdim], amS[Mdim], usS[Hdim], b1uS[Mdim], enS[CH];
    __shared__ float awS, ewS;

    for (int i = tid; i < Mdim * Hdim; i += 256) { W1s[i] = mW1[i]; W2s[i] = mW2[i]; }
    if (tid < Mdim) b1s[tid] = mb1[tid];
    if (tid < Hdim) b2s[tid] = mb2[tid];
    __threadfence();
    __syncthreads();

    for (int c = 0; c < NCHK; c++) {
        const int rowbase = b * Sseq + c * CH;

        // Phase 1: inter -> gelu -> act ; column means am
        {
            const int m = tid;
            const float* w1row = W1s + m * Hdim;
            float ama = 0.0f;
            for (int cc = 0; cc < CH; cc++) {
                const float* krow = kpost + (size_t)(rowbase + cc) * Hdim;
                float acc = b1s[m];
                for (int hh = 0; hh < Hdim; hh++) acc += krow[hh] * w1row[hh];
                float g = 0.5f * acc * (1.0f + erff(acc * 0.70710678f));
                act[cc][m] = (_Float16)g;
                ama += g;
            }
            amS[m] = ama * (1.0f / CH);
        }
        __syncthreads();

        // Phase 2: out = act @ W2^T + b2 ; err
        for (int jj = 0; jj < 32; jj++) {
            const int idx = tid + 256 * jj;          // 0..8191
            const int cc = idx >> 7, hh = idx & 127;
            const float* w2row = W2s + hh * Mdim;
            float acc = b2s[hh];
            for (int m = 0; m < Mdim; m++) acc += (float)act[cc][m] * w2row[m];
            combined[(size_t)(rowbase + cc) * Hdim + hh] = acc;
            float e = acc - vpost[(size_t)(rowbase + cc) * Hdim + hh];
            errh[cc][hh] = (_Float16)e;
        }
        __syncthreads();

        // Phase 3: per-row error norms
        if (tid < CH) {
            float s2 = 0.0f;
            for (int hh = 0; hh < Hdim; hh++) { float e = (float)errh[tid][hh]; s2 += e * e; }
            enS[tid] = sqrtf(s2);
        }
        __syncthreads();

        // Phase 4: grad clip by delta, column means us; aw/ew; zero b1u
        b1uS[tid] = 0.0f;
        if (tid < Hdim) {
            const int hh = tid;
            float usa = 0.0f;
            for (int cc = 0; cc < CH; cc++) {
                float e  = (float)errh[cc][hh];
                float d  = eda[(size_t)(rowbase + cc) * 384 + 128 + hh]; // delta
                float en = enS[cc];
                float g  = (en > d) ? d * e / (en + 1e-9f) : e;
                errh[cc][hh] = (_Float16)g;
                usa += g;
            }
            usS[hh] = usa * (1.0f / CH);
        }
        if (tid == 0) {
            const size_t rl = (size_t)(rowbase + CH - 1) * 384;
            float aw = 0.0f, ew = 0.0f;
            for (int hh = 0; hh < Hdim; hh++) { aw += eda[rl + 256 + hh]; ew += eda[rl + hh]; }
            awS = aw * (1.0f / Hdim); ewS = ew * (1.0f / Hdim);
        }
        __syncthreads();

        // Phase 5: rank-1 state updates; accumulate b1u[m] = sum_h us[h]*nW2[h][m]
        const float aw = awS, ew = ewS;
        for (int j = 0; j < 128; j++) {
            const int idx = tid + 256 * j;   // 0..32767
            {
                const int m = idx >> 7, hh = idx & 127;
                W1s[idx] = aw * W1s[idx] - ew * amS[m] * usS[hh];
            }
            {
                const int hh = idx >> 8, m = idx & 255;
                float nw2 = aw * W2s[idx] - ew * usS[hh] * amS[m];
                W2s[idx] = nw2;
                atomicAdd(&b1uS[m], usS[hh] * nw2);
            }
        }
        __syncthreads();

        // Phase 6: bias updates (each thread owns its slot)
        if (tid < Hdim) {
            const size_t rl = (size_t)(rowbase + CH - 1) * 384;
            float ab = eda[rl + 256 + tid], eb = eda[rl + tid];
            b2s[tid] = ab * b2s[tid] - eb * usS[tid];
        }
        b1s[tid] = aw * b1s[tid] - ew * b1uS[tid];
        __threadfence();
        __syncthreads();
    }
}

// ---------------------------------------------------------------------------
// Host launcher
// ---------------------------------------------------------------------------
extern "C" void kernel_launch(void* const* d_in, const int* in_sizes, int n_in,
                              void* d_out, int out_size, void* d_ws, size_t ws_size,
                              hipStream_t stream)
{
    (void)in_sizes; (void)n_in; (void)out_size; (void)ws_size;
    const float* x   = (const float*)d_in[0];
    const float* Wk  = (const float*)d_in[2];
    const float* Wv  = (const float*)d_in[3];
    const float* Wo  = (const float*)d_in[4];
    const float* Wg  = (const float*)d_in[5];
    const float* We1 = (const float*)d_in[6];
    const float* We2 = (const float*)d_in[7];
    const float* Wd1 = (const float*)d_in[8];
    const float* Wd2 = (const float*)d_in[9];
    const float* Wa1 = (const float*)d_in[10];
    const float* Wa2 = (const float*)d_in[11];
    const float* cw  = (const float*)d_in[12];
    const float* cb  = (const float*)d_in[13];
    const float* kng = (const float*)d_in[16];
    const float* knb = (const float*)d_in[17];
    const float* mW1 = (const float*)d_in[18];
    const float* mb1 = (const float*)d_in[19];
    const float* mW2 = (const float*)d_in[20];
    const float* mb2 = (const float*)d_in[21];
    float* out = (float*)d_out;

    // Workspace carve-up (256B aligned)
    char* ws = (char*)d_ws;
    size_t off = 0;
    auto carve = [&](size_t bytes) -> char* {
        char* p = ws + off; off += (bytes + 255) & ~(size_t)255; return p;
    };
    _Float16* x_h    = (_Float16*)carve((size_t)ROWS * Ddim * 2);   // 32MB
    _Float16* Wkv_t  = (_Float16*)carve((size_t)256 * Ddim * 2);    // N x K
    _Float16* Wp1_t  = (_Float16*)carve((size_t)256 * Ddim * 2);
    _Float16* B2_t   = (_Float16*)carve((size_t)384 * 256 * 2);
    _Float16* Wg_t   = (_Float16*)carve((size_t)Ddim * Ddim * 2);
    _Float16* Wo_t   = (_Float16*)carve((size_t)Ddim * Hdim * 2);
    float*    kvraw  = (float*)carve((size_t)ROWS * 256 * 4);       // 16MB
    float*    rraw   = (float*)carve((size_t)ROWS * 256 * 4);       // 16MB
    _Float16* r_h    = (_Float16*)carve((size_t)ROWS * 256 * 2);    // 8MB
    float*    eda    = (float*)carve((size_t)ROWS * 384 * 4);       // 24MB
    float*    kpost  = (float*)carve((size_t)ROWS * Hdim * 4);      // 8MB
    float*    vpost  = (float*)carve((size_t)ROWS * Hdim * 4);      // 8MB
    float*    comb   = (float*)carve((size_t)ROWS * Hdim * 4);      // 8MB
    _Float16* comb_h = (_Float16*)carve((size_t)ROWS * Hdim * 2);   // 4MB
    float*    W1st   = (float*)carve((size_t)Bsz * Mdim * Hdim * 4);
    float*    W2st   = (float*)carve((size_t)Bsz * Mdim * Hdim * 4);

    const int T = 256;

    // 1) casts + packing (weights transposed to N x K)
    {
        int n = ROWS * Ddim;
        cvt_f32_f16<<<(n + T - 1) / T, T, 0, stream>>>(x, x_h, n);
    }
    pack_kv_t  <<<(256 * Ddim + T - 1) / T, T, 0, stream>>>(Wk, Wv, Wkv_t);
    pack_p1_t  <<<(256 * Ddim + T - 1) / T, T, 0, stream>>>(We1, Wd1, Wa1, Wp1_t);
    pack_diag_t<<<(384 * 256 + T - 1) / T, T, 0, stream>>>(We2, Wd2, Wa2, B2_t);
    {
        int n = Ddim * Ddim;   // Wg: (K=1024, N=1024) -> Wg_t (1024 x 1024)
        cvt_f32_f16_t<<<(n + T - 1) / T, T, 0, stream>>>(Wg, Wg_t, Ddim, Ddim);
        int n2 = Ddim * Hdim;  // Wo: (K=128, N=1024) -> Wo_t (1024 x 128)
        cvt_f32_f16_t<<<(n2 + T - 1) / T, T, 0, stream>>>(Wo, Wo_t, Hdim, Ddim);
    }

    // 2) kv_raw = x @ [Wk|Wv]      (16384 x 256, K=1024)
    gemm_f16_wmma<<<dim3(256 / TILE_N, ROWS / TILE_M), T, 0, stream>>>(
        x_h, Wkv_t, kvraw, ROWS, 256, Ddim, 0);

    // 3) r_raw = x @ [We1|Wd1|Wa1|0]  (16384 x 256, K=1024)
    gemm_f16_wmma<<<dim3(256 / TILE_N, ROWS / TILE_M), T, 0, stream>>>(
        x_h, Wp1_t, rraw, ROWS, 256, Ddim, 0);
    {
        int n = ROWS * 256;
        cvt_f32_f16<<<(n + T - 1) / T, T, 0, stream>>>(rraw, r_h, n);
    }

    // 4) eda_raw = r @ blockdiag[We2;Wd2;Wa2]  (16384 x 384, K=256)
    gemm_f16_wmma<<<dim3(384 / TILE_N, ROWS / TILE_M), T, 0, stream>>>(
        r_h, B2_t, eda, ROWS, 384, 256, 0);
    {
        int n = ROWS * 384;
        eda_act<<<(n + T - 1) / T, T, 0, stream>>>(eda, n);
    }

    // 5) depthwise conv + LN(k)
    conv_ln<<<ROWS, 128, 0, stream>>>(kvraw, cw, cb, kng, knb, kpost, vpost);

    // 6) sequential chunked scan -> combined
    scan_kernel<<<Bsz, T, 0, stream>>>(kpost, vpost, eda, mW1, mb1, mW2, mb2,
                                       W1st, W2st, comb);
    {
        int n = ROWS * Hdim;
        cvt_f32_f16<<<(n + T - 1) / T, T, 0, stream>>>(comb, comb_h, n);
    }

    // 7) gate = x @ Wg -> d_out (pre-activation)
    gemm_f16_wmma<<<dim3(Ddim / TILE_N, ROWS / TILE_M), T, 0, stream>>>(
        x_h, Wg_t, out, ROWS, Ddim, Ddim, 0);

    // 8) out = (combined @ Wo) * sigmoid(gate)   (fused epilogue reads d_out)
    gemm_f16_wmma<<<dim3(Ddim / TILE_N, ROWS / TILE_M), T, 0, stream>>>(
        comb_h, Wo_t, out, ROWS, Ddim, Hdim, 1);
}
